// Model_22239340659183
// MI455X (gfx1250) — matrix operand order repaired and checked by reading, not compiled, not run on hardware
//
#include <hip/hip_runtime.h>
#include <hip/hip_bf16.h>

typedef __attribute__((ext_vector_type(16))) __bf16 v16bf;
typedef __attribute__((ext_vector_type(8)))  float  v8f;
typedef __attribute__((ext_vector_type(4)))  unsigned u32x4;
typedef __attribute__((ext_vector_type(8)))  unsigned u32x8;

#define DDIM 1024
#define BT   32768   // B*T rows
#define TT   512
#define BB   64
#define JJ   400

// ---------- helpers ----------
__device__ __forceinline__ __bf16 f2bf(float f) {
    unsigned u = __builtin_bit_cast(unsigned, f);
    unsigned r = u + 0x7FFFu + ((u >> 16) & 1u);   // round-to-nearest-even
    unsigned short h = (unsigned short)(r >> 16);
    return __builtin_bit_cast(__bf16, h);
}

__device__ __forceinline__ unsigned lds_off(const void* p) {
    // generic pointer to LDS: low 32 bits are the LDS offset
    return (unsigned)(size_t)p;
}

// per-lane async global->LDS copy of 16B (tracked by ASYNCcnt)
__device__ __forceinline__ void async_g2l_b128(void* lds_dst, const void* gsrc) {
    asm volatile("global_load_async_to_lds_b128 %0, %1, off"
                 :: "v"(lds_off(lds_dst)), "v"(gsrc) : "memory");
}
__device__ __forceinline__ void wait_async0() {
    asm volatile("s_wait_asynccnt 0x0" ::: "memory");
}

__device__ __forceinline__ void store_ctile(float* __restrict__ dst, v8f acc,
                                            int row0, int col, int ld, float bias) {
#pragma unroll
    for (int r = 0; r < 8; ++r)
        dst[(size_t)(row0 + r) * ld + col] = acc[r] + bias;
}

// ---------- kernel 0: fp32 -> bf16 weight repack ----------
__global__ void prep_weights(const float* __restrict__ Wch,
                             __bf16* __restrict__ Whn, __bf16* __restrict__ Wxn) {
    int idx = blockIdx.x * blockDim.x + threadIdx.x;
    if (idx >= DDIM * 2 * DDIM) return;
    int n  = idx / (2 * DDIM);
    int k2 = idx % (2 * DDIM);
    __bf16 v = f2bf(Wch[idx]);
    if (k2 < DDIM) Whn[(size_t)n * DDIM + k2] = v;
    else           Wxn[(size_t)n * DDIM + (k2 - DDIM)] = v;
}

// ---------- kernel 1: fused gather + GEMM  XW = bf16(emb[cap]) @ Wx^T + b ----------
#define BM 128
#define BN 64
#define BK 32
#define NK (DDIM / BK)
#define ALD 40   // padded LDS row stride (bf16), 80B: 16B aligned, conflict-free
#define BLD 40

__global__ __launch_bounds__(256) void gemm_xw(
    const int*   __restrict__ cap,  const float* __restrict__ emb,
    const __bf16* __restrict__ Wxn, const float* __restrict__ bias,
    float* __restrict__ XW)
{
    __shared__ int     rowsel[BM];
    __shared__ __bf16  As[2][BM * ALD];   // [m][k], double buffered
    __shared__ __bf16  Bs[2][BN * BLD];   // [n][k], double buffered

    const int tid = threadIdx.x;
    const int m0  = blockIdx.y * BM;
    const int n0  = blockIdx.x * BN;

    if (tid < BM) rowsel[tid] = cap[m0 + tid];
    __syncthreads();

    const int wave = tid >> 5;          // wave32
    const int lane = tid & 31;
    const int lm   = lane & 15;
    const int hi   = lane >> 4;
    const int ka   = hi * 8;            // A fragment K base
    const int kb16 = hi * 16;           // unused; B fragment also uses ka
    const int wm   = wave & 3;          // 4 M-chunks of 32 rows
    const int wn   = wave >> 2;         // 2 N-chunks of 32 cols

    // staging lambdas
    const int sm   = tid >> 1, shalf = tid & 1;        // A: 128 rows x 2 halves
    const int sn   = tid >> 2, sc = tid & 3;           // B: 64 rows x 4 chunks of 8 bf16

    auto stageA = [&](int k0, __bf16* Ab) {
        const float* src = emb + (size_t)rowsel[sm] * DDIM + k0 + shalf * 16;
        __bf16* dst = Ab + sm * ALD + shalf * 16;
#pragma unroll
        for (int i = 0; i < 16; i += 4) {
            float4 f = *(const float4*)(src + i);
            dst[i + 0] = f2bf(f.x); dst[i + 1] = f2bf(f.y);
            dst[i + 2] = f2bf(f.z); dst[i + 3] = f2bf(f.w);
        }
    };
    auto stageB = [&](int k0, __bf16* Bb) {   // async DMA, no VGPR round-trip
        async_g2l_b128(Bb + sn * BLD + sc * 8,
                       Wxn + (size_t)(n0 + sn) * DDIM + k0 + sc * 8);
    };

    v8f acc00 = {}, acc01 = {}, acc10 = {}, acc11 = {};

    stageB(0, Bs[0]);
    stageA(0, As[0]);
    wait_async0();
    __syncthreads();

    for (int kb = 0; kb < NK; ++kb) {
        const int cur = kb & 1;
        if (kb + 1 < NK) {               // overlap next-stage copies with WMMA
            stageB((kb + 1) * BK, Bs[cur ^ 1]);
            stageA((kb + 1) * BK, As[cur ^ 1]);
        }

        const __bf16* pa0 = As[cur] + (wm * 32 + lm) * ALD;
        const __bf16* pa1 = As[cur] + (wm * 32 + 16 + lm) * ALD;
        const __bf16* pb0 = Bs[cur] + (wn * 32 + lm) * BLD + ka;
        const __bf16* pb1 = Bs[cur] + (wn * 32 + 16 + lm) * BLD + ka;
        v16bf a0, a1, b0, b1;
#pragma unroll
        for (int i = 0; i < 8; ++i) {
            a0[i] = pa0[ka + i];  a0[8 + i] = pa0[16 + ka + i];
            a1[i] = pa1[ka + i];  a1[8 + i] = pa1[16 + ka + i];
            b0[i] = pb0[i];       b0[8 + i] = pb0[16 + i];
            b1[i] = pb1[i];       b1[8 + i] = pb1[16 + i];
        }
        acc00 = __builtin_amdgcn_wmma_f32_16x16x32_bf16(false, a0, false, b0, (short)0, acc00, false, false);
        acc01 = __builtin_amdgcn_wmma_f32_16x16x32_bf16(false, a0, false, b1, (short)0, acc01, false, false);
        acc10 = __builtin_amdgcn_wmma_f32_16x16x32_bf16(false, a1, false, b0, (short)0, acc10, false, false);
        acc11 = __builtin_amdgcn_wmma_f32_16x16x32_bf16(false, a1, false, b1, (short)0, acc11, false, false);

        wait_async0();                   // own async for next stage complete
        __syncthreads();                 // everyone staged & done reading
    }

    const int c0 = n0 + wn * 32 + lm;
    const int c1 = c0 + 16;
    const float bi0 = bias[c0], bi1 = bias[c1];
    store_ctile(XW, acc00, m0 + wm * 32 +      hi * 8, c0, DDIM, bi0);
    store_ctile(XW, acc01, m0 + wm * 32 +      hi * 8, c1, DDIM, bi1);
    store_ctile(XW, acc10, m0 + wm * 32 + 16 + hi * 8, c0, DDIM, bi0);
    store_ctile(XW, acc11, m0 + wm * 32 + 16 + hi * 8, c1, DDIM, bi1);
}

// ---------- kernel 2: sequential recurrence h = h @ Wh^T + xw_t ----------
// 16 persistent WGs, one 64-column slab each. Wh slab pinned in LDS (async
// preload), xw_t tile DMA'd by the Tensor Data Mover each step, h re-staged
// as bf16 each step, software global barrier between steps.
#define KP 1032   // padded LDS row stride (bf16): 2064B, 16B aligned, conflict-free

__global__ __launch_bounds__(256) void recurrence(
    const __bf16* __restrict__ Whn, const float* __restrict__ XW,
    float* __restrict__ H0, float* __restrict__ H1,
    unsigned* __restrict__ bar)
{
    extern __shared__ __bf16 smem[];
    __bf16* Whs = smem;                          // [64][KP]  W_h column slab, [n][k]
    __bf16* Hs  = smem + 64 * KP;                // [64][KP]  h as bf16, [m][k]
    float*  XWs = (float*)(smem + 2 * 64 * KP);  // [64][64]  xw_t tile (TDM dest)

    const int tid = threadIdx.x;
    const int n0  = blockIdx.x * 64;
    const unsigned nWG = gridDim.x;

    // one-time async preload of the W_h slab into LDS (reused for all 512 steps)
    for (int c = tid; c < 64 * (DDIM / 8); c += 256) {
        int n  = c >> 7;
        int k8 = (c & 127) * 8;
        async_g2l_b128(Whs + n * KP + k8, Whn + (size_t)(n0 + n) * DDIM + k8);
    }
    wait_async0();

    const int wave = tid >> 5, lane = tid & 31;
    const int lm = lane & 15, hi = lane >> 4;
    const int ka = hi * 8, kb16 = hi * 16;
    const int mt  = wave & 3;             // M tile (rows mt*16..)
    const int nt0 = (wave >> 2) * 2;      // two N tiles within slab

    for (int t = 0; t < TT; ++t) {
        const float* hsrc = (t & 1) ? H1 : H0;
        float*       hdst = (t & 1) ? H0 : H1;

        __syncthreads();

        // TDM: DMA xw[:, t, n0:n0+64] (64 rows x 256B, row stride 2MB) into LDS
        if (wave == 0) {
            unsigned long long ga = (unsigned long long)(size_t)(XW + (size_t)t * DDIM + n0);
            u32x4 g0;
            g0[0] = 1u;                                   // count=1, user desc
            g0[1] = lds_off(XWs);                         // lds_addr
            g0[2] = (unsigned)(ga & 0xFFFFFFFFu);         // global_addr[31:0]
            g0[3] = (unsigned)((ga >> 32) & 0x01FFFFFFu)  // global_addr[56:32]
                    | (2u << 30);                         // type = 2 (image)
            u32x8 g1;
            g1[0] = (3u << 16);          // workgroup_mask=0, data_size=3 (8B units)
            g1[1] = (32u << 16);         // tensor_dim0 = 32 units (256B tile row)
            g1[2] = (64u << 16);         // tensor_dim1 = 64 rows
            g1[3] = (32u << 16);         // tile_dim0 = 32 units
            g1[4] = 64u;                 // tile_dim1 = 64, tile_dim2 = 0
            g1[5] = (unsigned)((size_t)TT * DDIM * 4 / 8);  // dim0 stride = 262144 units
            g1[6] = 0u;
            g1[7] = 0u;
            asm volatile("tensor_load_to_lds %0, %1" :: "s"(g0), "s"(g1) : "memory");
        }

        // stage full h (64x1024 fp32 -> bf16) into LDS
        for (int c = tid; c < BB * (DDIM / 4); c += 256) {
            int m  = c >> 8;
            int k4 = (c & 255) * 4;
            float4 f = *(const float4*)(hsrc + (size_t)m * DDIM + k4);
            __bf16* d = Hs + m * KP + k4;
            d[0] = f2bf(f.x); d[1] = f2bf(f.y); d[2] = f2bf(f.z); d[3] = f2bf(f.w);
        }
        if (wave == 0) __builtin_amdgcn_s_wait_tensorcnt(0);
        __syncthreads();

        v8f acc0 = {}, acc1 = {};
        const __bf16* pa  = Hs  + (mt * 16 + lm) * KP;
        const __bf16* pb0 = Whs + ((nt0    ) * 16 + lm) * KP;
        const __bf16* pb1 = Whs + ((nt0 + 1) * 16 + lm) * KP;
        for (int k0 = 0; k0 < DDIM; k0 += 32) {
            v16bf a, b0, b1;
#pragma unroll
            for (int i = 0; i < 8; ++i) {
                a[i]      = pa[k0 + ka + i];
                a[8 + i]  = pa[k0 + 16 + ka + i];
                b0[i]     = pb0[k0 + ka + i];
                b0[8 + i] = pb0[k0 + 16 + ka + i];
                b1[i]     = pb1[k0 + ka + i];
                b1[8 + i] = pb1[k0 + 16 + ka + i];
            }
            acc0 = __builtin_amdgcn_wmma_f32_16x16x32_bf16(false, a, false, b0, (short)0, acc0, false, false);
            acc1 = __builtin_amdgcn_wmma_f32_16x16x32_bf16(false, a, false, b1, (short)0, acc1, false, false);
        }

        // epilogue: add xw tile (from LDS) and write h_next (fp32)
        {
            const int lcA = nt0 * 16 + lm;      // col within slab
            const int lcB = lcA + 16;
#pragma unroll
            for (int r = 0; r < 8; ++r) {
                int m = mt * 16 + hi * 8 + r;
                hdst[(size_t)m * DDIM + n0 + lcA] = acc0[r] + XWs[m * 64 + lcA];
                hdst[(size_t)m * DDIM + n0 + lcB] = acc1[r] + XWs[m * 64 + lcB];
            }
        }

        // global barrier across the 16 WGs
        __threadfence();
        __syncthreads();
        if (tid == 0) {
            atomicAdd(bar, 1u);
            unsigned target = nWG * (unsigned)(t + 1);
            while (*((volatile unsigned*)bar) < target)
                __builtin_amdgcn_s_sleep(1);
        }
        __syncthreads();
        __threadfence();
    }
}

// ---------- kernel 3: fusion ----------
__global__ void fusion(const float* __restrict__ x_img, const float* __restrict__ h,
                       const float* __restrict__ Wimg,  const float* __restrict__ bimg,
                       const float* __restrict__ Wcap,  const float* __restrict__ bcap,
                       float* __restrict__ XF)
{
    int idx = blockIdx.x * blockDim.x + threadIdx.x;
    if (idx >= BB * JJ) return;
    int m = idx / JJ, j = idx % JJ;
    float s = bimg[j] + bcap[j];
    const float* xi = x_img + (size_t)m * 512;
    const float* wi = Wimg  + (size_t)j * 512;
    for (int k = 0; k < 512; ++k)  s += xi[k] * wi[k];
    const float* hh = h    + (size_t)m * DDIM;
    const float* wc = Wcap + (size_t)j * DDIM;
    for (int k = 0; k < DDIM; ++k) s += hh[k] * wc[k];
    XF[idx] = s;
}

// ---------- kernel 4: three tiny heads ----------
__global__ void heads(const float* __restrict__ XF,
                      const float* __restrict__ Wint, const float* __restrict__ bint,
                      const float* __restrict__ Wsem, const float* __restrict__ bsem,
                      const float* __restrict__ Wctx, const float* __restrict__ bctx,
                      float* __restrict__ out)
{
    int o = blockIdx.x * blockDim.x + threadIdx.x;
    if (o >= BB * 13) return;
    const float *W, *bsel; int m, j;
    if (o < BB * 7)       { m = o / 7;            j = o % 7;  W = Wint; bsel = bint; }
    else if (o < BB * 10) { int o2 = o - BB * 7;  m = o2 / 3; j = o2 % 3; W = Wsem; bsel = bsem; }
    else                  { int o3 = o - BB * 10; m = o3 / 3; j = o3 % 3; W = Wctx; bsel = bctx; }
    float s = bsel[j];
    const float* xf = XF + (size_t)m * JJ;
    const float* w  = W  + (size_t)j * JJ;
    for (int k = 0; k < JJ; ++k) s += xf[k] * w[k];
    out[o] = s;
}

// ---------- launch ----------
extern "C" void kernel_launch(void* const* d_in, const int* in_sizes, int n_in,
                              void* d_out, int out_size, void* d_ws, size_t ws_size,
                              hipStream_t stream)
{
    const float* x_img = (const float*)d_in[0];
    const int*   cap   = (const int*)  d_in[1];
    const float* emb   = (const float*)d_in[2];
    const float* Wch   = (const float*)d_in[3];
    const float* bch   = (const float*)d_in[4];
    const float* Wimg  = (const float*)d_in[5];
    const float* bimg  = (const float*)d_in[6];
    const float* Wcap  = (const float*)d_in[7];
    const float* bcap  = (const float*)d_in[8];
    const float* Wint  = (const float*)d_in[9];
    const float* bint  = (const float*)d_in[10];
    const float* Wsem  = (const float*)d_in[11];
    const float* bsem  = (const float*)d_in[12];
    const float* Wctx  = (const float*)d_in[13];
    const float* bctx  = (const float*)d_in[14];
    float* out = (float*)d_out;

    char* ws = (char*)d_ws;
    size_t off = 0;
    __bf16* Wxn = (__bf16*)(ws + off); off += (size_t)DDIM * DDIM * 2;   // 2 MB
    __bf16* Whn = (__bf16*)(ws + off); off += (size_t)DDIM * DDIM * 2;   // 2 MB
    float*  XW  = (float*) (ws + off); off += (size_t)BT * DDIM * 4;     // 128 MB
    float*  H0  = (float*) (ws + off); off += (size_t)BB * DDIM * 4;
    float*  H1  = (float*) (ws + off); off += (size_t)BB * DDIM * 4;
    float*  XF  = (float*) (ws + off); off += (size_t)BB * JJ * 4;
    unsigned* bar = (unsigned*)(ws + off); off += 256;

    hipMemsetAsync(H0, 0, (size_t)BB * DDIM * 4, stream);   // h0 = zeros
    hipMemsetAsync((void*)bar, 0, 256, stream);             // barrier counter

    prep_weights<<<(DDIM * 2 * DDIM + 255) / 256, 256, 0, stream>>>(Wch, Whn, Wxn);

    dim3 g1(DDIM / BN, BT / BM);   // (16, 256)
    gemm_xw<<<g1, 256, 0, stream>>>(cap, emb, Wxn, bch, XW);

    size_t smem = (size_t)2 * 64 * KP * sizeof(__bf16) + 64 * 64 * sizeof(float);
    recurrence<<<16, 256, smem, stream>>>(Whn, XW, H0, H1, bar);

    fusion<<<(BB * JJ + 255) / 256, 256, 0, stream>>>(x_img, H0, Wimg, bimg, Wcap, bcap, XF);
    heads<<<(BB * 13 + 255) / 256, 256, 0, stream>>>(XF, Wint, bint, Wsem, bsem, Wctx, bctx, out);
}
